// AdaptiveLayer_65429531787287
// MI455X (gfx1250) — matrix operand order repaired and checked by reading, not compiled, not run on hardware
//
#include <hip/hip_runtime.h>
#include <hip/hip_bf16.h>

// ---------------------------------------------------------------------------
// Sinkhorn memory-assignment layer for MI455X (gfx1250, wave32, WMMA).
//
// Pipeline (per-column Sinkhorn scalars cancel in the argmax; per-row
// factors telescope: r1*r2 = t2, r1*r2*r3 = t3):
//   1) l2-normalize tokens & memories -> f16
//   2) WMMA f16 GEMM: A[n][m] = exp((xn.mn)/0.05), B-tile staged via
//      global_load_async_to_lds_b128 (double-buffered); fused t1[m]=sum_n A
//   3) w1[n] = 1 / sum_m A/t1        (token pass)
//   4) t2[m] = sum_n A*w1            (column pass)
//   5) w2[n] = 1 / sum_m A/t2        (token pass)
//   6) t3[m] = sum_n A*w2            (column pass)
//   7) idx[n] = argmax_m A[n][m]/t3[m]; out = 0.5*(proj + memory[idx])
//
// Score matrix A (128 MB fp32) is kept L2-resident (192 MB L2).
// Workspace: N*M*4 + N*D*2 + M*D*2 + (N+2M)*4 ~= 204 MB.
// ---------------------------------------------------------------------------

typedef __attribute__((ext_vector_type(16))) _Float16 v16h;
typedef __attribute__((ext_vector_type(8)))  _Float16 v8h;
typedef __attribute__((ext_vector_type(8)))  float    v8f;

#define DDIM 1024

// ---------------- row L2-normalize, f32 -> f16 -----------------------------
__global__ __launch_bounds__(256) void l2norm_rows_f16(
    const float* __restrict__ src, _Float16* __restrict__ dst, int D) {
  const int r = blockIdx.x, tid = threadIdx.x;
  const float* row = src + (size_t)r * D;
  float ss = 0.f;
  for (int d = tid; d < D; d += 256) { float v = row[d]; ss += v * v; }
  for (int off = 16; off; off >>= 1) ss += __shfl_down(ss, off, 32);
  __shared__ float red[8];
  if ((tid & 31) == 0) red[tid >> 5] = ss;
  __syncthreads();
  if (tid < 8) {
    float v = red[tid];
    v += __shfl_down(v, 4, 32);
    v += __shfl_down(v, 2, 32);
    v += __shfl_down(v, 1, 32);
    if (tid == 0) red[0] = v;
  }
  __syncthreads();
  const float inv = rsqrtf(fmaxf(red[0], 1e-12f));
  for (int d = tid; d < D; d += 256)
    dst[(size_t)r * D + d] = (_Float16)(row[d] * inv);
}

// ---------------- f16 A-fragment load (16x32, contiguous k = 16*hi + s) ----------
__device__ __forceinline__ v16h load_a_frag(const _Float16* rowbase, int d0, int hi) {
  return *(const v16h*)(rowbase + d0 + hi * 16);
}

// ---------------- WMMA GEMM + exp(sim/T) + fused column sums ----------------
// Block = 256 threads (8 wave32), tile = 128 tokens x 128 mems.
// B tile (128 mems x 32 d, 8 KB) is async-staged into LDS, double-buffered.
__global__ __launch_bounds__(256) void sim_exp_gemm(
    const _Float16* __restrict__ xh, const _Float16* __restrict__ mh,
    float* __restrict__ A, float* __restrict__ t1, int M, int D) {
  const int tid  = threadIdx.x;
  const int lane = tid & 31;
  const int wave = tid >> 5;
  const int wv_n = wave & 3;          // 4 token groups of 32
  const int wv_m = wave >> 2;         // 2 mem groups of 64
  const int lo = lane & 15, hi = lane >> 4;

  const int n0 = blockIdx.x * 128 + wv_n * 32;   // token base of this wave
  const int mtile = blockIdx.y * 128;            // block mem base
  const int m0loc = wv_m * 64;                   // wave mem base inside tile

  // double-buffered B tile: [buf][row(mem) 0..127][d 0..31] f16, row = 64 B
  __shared__ _Float16 Bt[2][128][32];

  // issue 2 x b128 async loads per thread: 512 chunks of 16 B cover 8 KB
  auto issue_b = [&](int buf, int d0) {
    const unsigned ldsbase = (unsigned)(size_t)&Bt[buf][0][0];
#pragma unroll
    for (int j = 0; j < 2; ++j) {
      const int c   = tid * 2 + j;
      const int row = c >> 2;          // 0..127 (mem row in tile)
      const int col = c & 3;           // 16-byte chunk within the 64 B row
      const unsigned voff  = (unsigned)(((mtile + row) * D + d0 + col * 8) * 2);
      const unsigned laddr = ldsbase + (unsigned)(row * 64 + col * 16);
      asm volatile("global_load_async_to_lds_b128 %0, %1, %2"
                   :: "v"(laddr), "v"(voff), "s"(mh) : "memory");
    }
  };

  v8f zero = {};
  v8f acc[2][4];
#pragma unroll
  for (int i = 0; i < 2; ++i)
#pragma unroll
    for (int j = 0; j < 4; ++j) acc[i][j] = zero;

  const _Float16* arow = xh + (size_t)(n0 + lo) * D;  // + tn*16*D per subtile

  issue_b(0, 0);
  int buf = 0;
  for (int d0 = 0; d0 < D; d0 += 32) {
    asm volatile("s_wait_asynccnt 0x0" ::: "memory");
    __syncthreads();
    if (d0 + 32 < D) issue_b(buf ^ 1, d0 + 32);

    v16h af[2], bfr[4];
#pragma unroll
    for (int tn = 0; tn < 2; ++tn)
      af[tn] = load_a_frag(arow + (size_t)tn * 16 * D, d0, hi);
#pragma unroll
    for (int tm = 0; tm < 4; ++tm) {
      // B 32x16 frag: lane holds col n = lane&15, K = 16*(lane>>4)+e
      bfr[tm] = *(const v16h*)&Bt[buf][m0loc + tm * 16 + lo][hi * 16];
    }
#pragma unroll
    for (int tn = 0; tn < 2; ++tn)
#pragma unroll
      for (int tm = 0; tm < 4; ++tm)
        acc[tn][tm] = __builtin_amdgcn_wmma_f32_16x16x32_f16(
            false, af[tn], false, bfr[tm], (short)0, acc[tn][tm], false, false);
    buf ^= 1;
  }

  // epilogue: exp in-place, store A, fused partial column sums -> t1
  const int m0 = mtile + m0loc;
#pragma unroll
  for (int tn = 0; tn < 2; ++tn) {
#pragma unroll
    for (int tm = 0; tm < 4; ++tm) {
      const int mem = m0 + tm * 16 + lo;
#pragma unroll
      for (int v = 0; v < 8; ++v) {
        const float e = __expf(acc[tn][tm][v] * 20.0f);  // 1/T = 20
        acc[tn][tm][v] = e;
        const int tok = n0 + tn * 16 + v + 8 * hi;
        A[(size_t)tok * M + mem] = e;
      }
    }
  }
#pragma unroll
  for (int tm = 0; tm < 4; ++tm) {
    float cs = 0.f;
#pragma unroll
    for (int tn = 0; tn < 2; ++tn)
#pragma unroll
      for (int v = 0; v < 8; ++v) cs += acc[tn][tm][v];
    cs += __shfl_xor(cs, 16, 32);      // lanes l and l^16 share the mem col
    if (lane < 16) atomicAdd(&t1[m0 + tm * 16 + lane], cs);
  }
}

// ---------------- small vector helpers -------------------------------------
__global__ void zero_vec(float* __restrict__ t, int n) {
  int i = blockIdx.x * blockDim.x + threadIdx.x;
  if (i < n) t[i] = 0.f;
}
__global__ void recip_vec(const float* __restrict__ t, float* __restrict__ g, int n) {
  int i = blockIdx.x * blockDim.x + threadIdx.x;
  if (i < n) g[i] = 1.0f / t[i];
}

// ---------------- column pass: t[m] += sum_n A[n][m] * w[n] -----------------
__global__ __launch_bounds__(256) void col_weighted_sum(
    const float* __restrict__ A, const float* __restrict__ w,
    float* __restrict__ t, int M) {
  const int tid = threadIdx.x;
  const int nbase = blockIdx.x * 256;
  float a0 = 0.f, a1 = 0.f, a2 = 0.f, a3 = 0.f;
  for (int i = 0; i < 256; ++i) {
    const int n = nbase + i;
    const float wn = w[n];
    const float* row = A + (size_t)n * M;
    __builtin_prefetch(row + 8 * M + tid, 0, 0);   // global_prefetch_b8
    a0 += row[tid      ] * wn;
    a1 += row[tid + 256] * wn;
    a2 += row[tid + 512] * wn;
    a3 += row[tid + 768] * wn;
  }
  atomicAdd(&t[tid      ], a0);
  atomicAdd(&t[tid + 256], a1);
  atomicAdd(&t[tid + 512], a2);
  atomicAdd(&t[tid + 768], a3);
}

// ---------------- token pass: out[n] = 1 / sum_m A[n][m] * g[m] -------------
__global__ __launch_bounds__(256) void token_weighted_recip_dot(
    const float* __restrict__ A, const float* __restrict__ g,
    float* __restrict__ out, int M) {
  const int lane = threadIdx.x & 31;
  const int wv = threadIdx.x >> 5;
  const int n = blockIdx.x * 8 + wv;
  const float* row = A + (size_t)n * M;
  float s = 0.f;
  for (int i = lane; i < M; i += 32) s += row[i] * g[i];
  for (int off = 16; off; off >>= 1) s += __shfl_down(s, off, 32);
  if (lane == 0) out[n] = 1.0f / s;
}

// ---------------- final argmax + gather + average ---------------------------
__global__ __launch_bounds__(256) void finalize_assign(
    const float* __restrict__ A, const float* __restrict__ g,
    const float* __restrict__ proj, const float* __restrict__ memry,
    float* __restrict__ outp, int M, int D) {
  const int n = blockIdx.x, tid = threadIdx.x;
  const float* row = A + (size_t)n * M;
  float best = -1.0f; int bidx = 0x7FFFFFFF;
  for (int i = tid; i < M; i += 256) {
    const float v = row[i] * g[i];
    if (v > best) { best = v; bidx = i; }
  }
  __shared__ float sv[256];
  __shared__ int   si[256];
  sv[tid] = best; si[tid] = bidx;
  __syncthreads();
  for (int s = 128; s; s >>= 1) {
    if (tid < s) {
      const float ov = sv[tid + s]; const int oi = si[tid + s];
      if (ov > sv[tid] || (ov == sv[tid] && oi < si[tid])) { sv[tid] = ov; si[tid] = oi; }
    }
    __syncthreads();
  }
  const int idx = si[0];
  const float* mrow = memry + (size_t)idx * D;
  const float* prow = proj + (size_t)n * D;
  float* orow = outp + (size_t)n * D;
  for (int d = tid; d < D; d += 256)
    orow[d] = 0.5f * (prow[d] + mrow[d]);
}

// ---------------------------------------------------------------------------
extern "C" void kernel_launch(void* const* d_in, const int* in_sizes, int n_in,
                              void* d_out, int out_size, void* d_ws, size_t ws_size,
                              hipStream_t stream) {
  const float* proj  = (const float*)d_in[0];   // [B,T,D] f32
  const float* memry = (const float*)d_in[1];   // [M,D]   f32
  const int D = DDIM;
  const int N = in_sizes[0] / D;   // 32768 tokens
  const int M = in_sizes[1] / D;   // 1024 memories

  // workspace carve-up
  char* ws = (char*)d_ws;
  float*    Amat = (float*)ws;                              // N*M f32 (128 MB)
  _Float16* xh   = (_Float16*)(ws + (size_t)N * M * 4);     // N*D f16
  _Float16* mh   = xh + (size_t)N * D;                      // M*D f16
  float*    wbuf = (float*)(mh + (size_t)M * D);            // N f32
  float*    tvec = wbuf + N;                                // M f32
  float*    gvec = tvec + M;                                // M f32

  float* outp = (float*)d_out;
  const int mb = (M + 255) / 256;

  // 1) normalize
  l2norm_rows_f16<<<N, 256, 0, stream>>>(proj, xh, D);
  l2norm_rows_f16<<<M, 256, 0, stream>>>(memry, mh, D);

  // 2) WMMA GEMM + exp -> A, fused t1[m] = sum_n A (L2-resident matrix)
  zero_vec<<<mb, 256, 0, stream>>>(tvec, M);
  sim_exp_gemm<<<dim3(N / 128, M / 128), 256, 0, stream>>>(xh, mh, Amat, tvec, M, D);
  recip_vec<<<mb, 256, 0, stream>>>(tvec, gvec, M);

  // 3) w1[n] = 1 / sum_m A*g
  token_weighted_recip_dot<<<N / 8, 256, 0, stream>>>(Amat, gvec, wbuf, M);

  // 4) t2[m] = sum_n A*w1 ; g = 1/t2
  zero_vec<<<mb, 256, 0, stream>>>(tvec, M);
  col_weighted_sum<<<N / 256, 256, 0, stream>>>(Amat, wbuf, tvec, M);
  recip_vec<<<mb, 256, 0, stream>>>(tvec, gvec, M);

  // 5) w2[n] = 1 / sum_m A*g
  token_weighted_recip_dot<<<N / 8, 256, 0, stream>>>(Amat, gvec, wbuf, M);

  // 6) t3[m] = sum_n A*w2 ; g = 1/t3  (t3 == r1*r2*r3)
  zero_vec<<<mb, 256, 0, stream>>>(tvec, M);
  col_weighted_sum<<<N / 256, 256, 0, stream>>>(Amat, wbuf, tvec, M);
  recip_vec<<<mb, 256, 0, stream>>>(tvec, gvec, M);

  // 7) argmax + gather + average
  finalize_assign<<<N, 256, 0, stream>>>(Amat, gvec, proj, memry, outp, M, D);
}
